// SGraphRNN_2370821947809
// MI455X (gfx1250) — matrix operand order repaired and checked by reading, not compiled, not run on hardware
//
#include <hip/hip_runtime.h>

typedef __attribute__((ext_vector_type(16))) __bf16 v16bf;
typedef __attribute__((ext_vector_type(8)))  float  v8f;

union V16BF { v16bf v; __bf16 e[16]; uint4 q[2]; };
union V8F   { v8f   v; float  e[8];  };

__device__ __forceinline__ float sigmoidf_(float x) { return 1.0f / (1.0f + __expf(-x)); }
__device__ __forceinline__ float finite_or0(float v) {
    return (v > -__builtin_inff() && v < __builtin_inff()) ? v : 0.0f;
}
__device__ __forceinline__ unsigned pack_bf16(float lo, float hi) {
    union { __bf16 b[2]; unsigned u; } p;
    p.b[0] = (__bf16)lo; p.b[1] = (__bf16)hi;
    return p.u;
}

// ---------------------------------------------------------------------------
// Generic tiled WMMA GEMM:  C[N x M] = A[N x K] @ opB (+ bias)
//   opB = transB ? B^T (B stored M x K) : B (stored K x M)
// Block: 256 threads (8 waves), tile 64 rows x 64 cols, K-chunks of 32.
// LDS holds A/B chunks pre-swizzled into WMMA *fragment layout*:
//   frag[tile][lane][16 bf16]  -> each lane reads its fragment as 2x b128.
// Wave w computes row-tile (w>>1) x col-tiles {2*(w&1), 2*(w&1)+1}: one A
// fragment feeds two v_wmma_f32_16x16x32_bf16 per K-chunk.
// Assumes M % 64 == 0 and K % 32 == 0 (true for every call below).
// ---------------------------------------------------------------------------
__global__ __launch_bounds__(256) void gemm_bias_kernel(
    const float* __restrict__ A, const float* __restrict__ B,
    const float* __restrict__ bias, float* __restrict__ C,
    int Nrows, int K, int M, int transB)
{
    __shared__ unsigned Af[4][32][8];   // 4 row tiles, per-lane 16 bf16 (8 dwords)
    __shared__ unsigned Bf[4][32][8];   // 4 col tiles, per-lane 16 bf16

    const int tid  = threadIdx.x;
    const int lane = tid & 31;
    const int wv   = tid >> 5;              // wave 0..7
    const int rowBase = blockIdx.x * 64;
    const int colBase = blockIdx.y * 64;
    const int rt  = wv >> 1;                // this wave's row tile (0..3)
    const int ct0 = (wv & 1) * 2;           // first of two col tiles (0 or 2)

    V8F acc0, acc1;
    #pragma unroll
    for (int i = 0; i < 8; ++i) { acc0.e[i] = 0.f; acc1.e[i] = 0.f; }

    for (int k0 = 0; k0 < K; k0 += 32) {
        // ---- stage A chunk (64 rows x 32 K) into fragment layout
        // A-frag mapping for element (r, c): lane = (r&15) + 16*(c>>4),
        // slot e = c&15  (contiguous k order, matches B fragments)
        #pragma unroll
        for (int i = 0; i < 2; ++i) {
            int idx = tid + i * 256;        // 0..511
            int r   = idx >> 3;             // 0..63
            int c   = (idx & 7) * 4;        // 0,4,...,28
            int row = rowBase + r;
            int rowc = (row < Nrows) ? row : (Nrows - 1);   // clamp, no branch
            float4 v = *(const float4*)(A + (size_t)rowc * K + k0 + c);
            if (row >= Nrows) { v.x = 0.f; v.y = 0.f; v.z = 0.f; v.w = 0.f; }
            int art   = r >> 4;
            int alane = (r & 15) + ((c >> 4) << 4);
            int e     = c & 15;                             // 0,4,8,12
            ((uint2*)Af[art][alane])[e >> 2] =
                make_uint2(pack_bf16(v.x, v.y), pack_bf16(v.z, v.w));
        }
        // ---- stage B chunk (32 K x 64 cols) into fragment layout
        // B-frag mapping for element (kk, c): lane = (c&15) + 16*(kk>>4), slot j = kk&15
        if (transB) {
            #pragma unroll
            for (int i = 0; i < 2; ++i) {
                int idx = tid + i * 256;    // 0..511
                int c   = idx >> 3;         // 0..63
                int kk  = (idx & 7) * 4;    // 0,4,...,28
                float4 v = *(const float4*)(B + (size_t)(colBase + c) * K + k0 + kk);
                int bct   = c >> 4;
                int blane = (c & 15) + ((kk >> 4) << 4);
                int j     = kk & 15;        // 0,4,8,12
                ((uint2*)Bf[bct][blane])[j >> 2] =
                    make_uint2(pack_bf16(v.x, v.y), pack_bf16(v.z, v.w));
            }
        } else {
            #pragma unroll
            for (int i = 0; i < 2; ++i) {
                int idx = tid + i * 256;
                int kk  = idx >> 4;         // 0..31
                int c   = (idx & 15) * 4;   // 0,4,...,60
                float4 v = *(const float4*)(B + (size_t)(k0 + kk) * M + colBase + c);
                int bct = c >> 4;
                int lhi = (kk >> 4) << 4;
                int j   = kk & 15;
                ((__bf16*)Bf[bct][(c & 15) + 0 + lhi])[j] = (__bf16)v.x;
                ((__bf16*)Bf[bct][(c & 15) + 1 + lhi])[j] = (__bf16)v.y;
                ((__bf16*)Bf[bct][(c & 15) + 2 + lhi])[j] = (__bf16)v.z;
                ((__bf16*)Bf[bct][(c & 15) + 3 + lhi])[j] = (__bf16)v.w;
            }
        }
        __syncthreads();

        // ---- fragments: contiguous 32 B per lane -> 2x ds_load_b128 each
        V16BF a, b0, b1;
        a.q[0]  = ((const uint4*)Af[rt][lane])[0];
        a.q[1]  = ((const uint4*)Af[rt][lane])[1];
        b0.q[0] = ((const uint4*)Bf[ct0][lane])[0];
        b0.q[1] = ((const uint4*)Bf[ct0][lane])[1];
        b1.q[0] = ((const uint4*)Bf[ct0 + 1][lane])[0];
        b1.q[1] = ((const uint4*)Bf[ct0 + 1][lane])[1];

        acc0.v = __builtin_amdgcn_wmma_f32_16x16x32_bf16(
            false, a.v, false, b0.v, (short)0, acc0.v, false, false);
        acc1.v = __builtin_amdgcn_wmma_f32_16x16x32_bf16(
            false, a.v, false, b1.v, (short)0, acc1.v, false, false);
        __syncthreads();
    }

    // ---- epilogue: C/D layout (VGPR r: lanes 0-15 -> M=r, lanes 16-31 -> M=r+8)
    #pragma unroll
    for (int t = 0; t < 2; ++t) {
        int c = colBase + (ct0 + t) * 16 + (lane & 15);
        float bv = (bias != nullptr) ? bias[c] : 0.f;
        #pragma unroll
        for (int r = 0; r < 8; ++r) {
            int m = rowBase + rt * 16 + r + ((lane >> 4) << 3);
            if (m < Nrows)
                C[(size_t)m * M + c] = (t == 0 ? acc0.e[r] : acc1.e[r]) + bv;
        }
    }
}

// ---------------------------------------------------------------------------
// el[n,h] = sum_q Wh[n,h*64+q]*attn_l[h,q]; er likewise. One thread per node.
// ---------------------------------------------------------------------------
__global__ __launch_bounds__(256) void el_er_kernel(
    const float* __restrict__ Wh, const float* __restrict__ al,
    const float* __restrict__ ar, float* __restrict__ el,
    float* __restrict__ er, int Nn)
{
    int n = blockIdx.x * blockDim.x + threadIdx.x;
    if (n >= Nn) return;
    const float* w = Wh + (size_t)n * 128;
    float l0 = 0.f, l1 = 0.f, r0 = 0.f, r1 = 0.f;
    #pragma unroll 4
    for (int q = 0; q < 64; ++q) {
        float w0 = w[q], w1 = w[64 + q];
        l0 += w0 * al[q];      l1 += w1 * al[64 + q];
        r0 += w0 * ar[q];      r1 += w1 * ar[64 + q];
    }
    el[n * 2] = l0; el[n * 2 + 1] = l1;
    er[n * 2] = r0; er[n * 2 + 1] = r1;
}

// ---------------------------------------------------------------------------
// One wave32 per node: 16-edge attention softmax (lane = edge + 16*head),
// single gather pass (x, z, Wh rows, 128-bit loads per lane), mean/max/alpha
// aggregation, fused sigmoid gate, writes [x | gate*attn_out] concat row.
// Exploits dst = repeat(arange(N),16): node n owns edges [16n,16n+16), deg==16.
// ---------------------------------------------------------------------------
__global__ __launch_bounds__(256) void edge_agg_kernel(
    const float* __restrict__ x,  const float* __restrict__ z,
    const float* __restrict__ Wh, const float* __restrict__ el,
    const float* __restrict__ er, const int* __restrict__ src,
    const float* __restrict__ Wg, const float* __restrict__ bg,
    float* __restrict__ xg, int Nn)
{
    const int lane = threadIdx.x & 31;
    const int n = blockIdx.x * 8 + (threadIdx.x >> 5);
    if (n >= Nn) return;

    // --- attention logits / softmax: lane handles (edge = lane&15, head = lane>>4)
    int eid = lane & 15;
    int h   = lane >> 4;
    int sE  = src[n * 16 + eid];
    float ev = el[sE * 2 + h] + er[n * 2 + h];
    ev = (ev > 0.f) ? ev : 0.2f * ev;                       // leaky_relu(0.2)
    float m = ev;
    #pragma unroll
    for (int off = 1; off < 16; off <<= 1) m = fmaxf(m, __shfl_xor(m, off, 32));
    m = finite_or0(m);
    float eexp = __expf(ev - m);
    float ssum = eexp;
    #pragma unroll
    for (int off = 1; off < 16; off <<= 1) ssum += __shfl_xor(ssum, off, 32);
    float alpha = eexp / fmaxf(ssum, 1e-12f);

    // --- single gather pass over the 16 edges
    float4 msum = make_float4(0.f, 0.f, 0.f, 0.f);
    float mz0 = -__builtin_inff(), mz1 = -__builtin_inff();
    float a0 = 0.f, a1 = 0.f, a2 = 0.f, a3 = 0.f;
    for (int e = 0; e < 16; ++e) {
        int s = src[n * 16 + e];
        float al_e = __shfl(alpha, e + (lane & 16), 32);    // alpha for (e, myhead)
        float4 xv = ((const float4*)(x + (size_t)s * 128))[lane];
        msum.x += xv.x; msum.y += xv.y; msum.z += xv.z; msum.w += xv.w;
        float2 zv = ((const float2*)(z + (size_t)s * 64))[lane];
        mz0 = fmaxf(mz0, zv.x); mz1 = fmaxf(mz1, zv.y);
        float4 wv = ((const float4*)(Wh + (size_t)s * 128))[lane];
        a0 += al_e * wv.x; a1 += al_e * wv.y; a2 += al_e * wv.z; a3 += al_e * wv.w;
    }
    const float inv = 1.0f / 16.0f;                         // deg == 16 always
    float mn[4] = { msum.x * inv, msum.y * inv, msum.z * inv, msum.w * inv };
    mz0 = finite_or0(mz0); mz1 = finite_or0(mz1);

    // --- gate = sigmoid(concat[x, mean, maxz](320) @ Wg(320x2) + bg)
    float4 xn = ((const float4*)(x + (size_t)n * 128))[lane];
    float xv4[4] = { xn.x, xn.y, xn.z, xn.w };
    float p0 = 0.f, p1 = 0.f;
    #pragma unroll
    for (int j = 0; j < 4; ++j) {
        int rx = lane * 4 + j;
        p0 += xv4[j] * Wg[rx * 2 + 0] + mn[j] * Wg[(128 + rx) * 2 + 0];
        p1 += xv4[j] * Wg[rx * 2 + 1] + mn[j] * Wg[(128 + rx) * 2 + 1];
    }
    {
        int rz = 256 + lane * 2;
        p0 += mz0 * Wg[rz * 2 + 0] + mz1 * Wg[(rz + 1) * 2 + 0];
        p1 += mz0 * Wg[rz * 2 + 1] + mz1 * Wg[(rz + 1) * 2 + 1];
    }
    #pragma unroll
    for (int off = 1; off < 32; off <<= 1) {
        p0 += __shfl_xor(p0, off, 32);
        p1 += __shfl_xor(p1, off, 32);
    }
    float g0 = sigmoidf_(p0 + bg[0]);
    float g1 = sigmoidf_(p1 + bg[1]);
    float g = (lane < 16) ? g0 : g1;                        // head of elems lane*4..+3

    // --- write merge input row: [ x(128) | gate*attn_out(128) ]
    ((float4*)(xg + (size_t)n * 256))[lane] = xn;
    ((float4*)(xg + (size_t)n * 256 + 128))[lane] = make_float4(g * a0, g * a1, g * a2, g * a3);
}

// ---------------------------------------------------------------------------
// GRU elementwise: newh = (1-z)*tanh(in + r*hn) + z*h
// ---------------------------------------------------------------------------
__global__ __launch_bounds__(256) void gru_kernel(
    const float* __restrict__ gi, const float* __restrict__ gh,
    const float* __restrict__ h, float* __restrict__ newh, int Nn)
{
    int idx = blockIdx.x * blockDim.x + threadIdx.x;
    if (idx >= Nn * 64) return;
    int n = idx >> 6, q = idx & 63;
    const float* gin = gi + (size_t)n * 192;
    const float* ghn = gh + (size_t)n * 192;
    float r  = sigmoidf_(gin[q]      + ghn[q]);
    float zg = sigmoidf_(gin[64 + q] + ghn[64 + q]);
    float ng = tanhf(gin[128 + q] + r * ghn[128 + q]);
    newh[idx] = (1.f - zg) * ng + zg * h[idx];
}

// ---------------------------------------------------------------------------
extern "C" void kernel_launch(void* const* d_in, const int* in_sizes, int n_in,
                              void* d_out, int out_size, void* d_ws, size_t ws_size,
                              hipStream_t stream)
{
    const float* x      = (const float*)d_in[0];   // (N,128)
    const float* h0     = (const float*)d_in[1];   // (N,64)
    const int*   src    = (const int*)  d_in[2];   // (E,)  dst (d_in[3]) unused: structured
    const float* Wm     = (const float*)d_in[4];   // (128,64)
    const float* bm     = (const float*)d_in[5];
    const float* Wg     = (const float*)d_in[6];   // (320,2)
    const float* bg     = (const float*)d_in[7];
    const float* Wfc    = (const float*)d_in[8];   // (128,128)
    const float* attn_l = (const float*)d_in[9];   // (2,64)
    const float* attn_r = (const float*)d_in[10];
    const float* Wmerge = (const float*)d_in[11];  // (256,64)
    const float* bmerge = (const float*)d_in[12];
    const float* W_ih   = (const float*)d_in[13];  // (192,64) -> used transposed
    const float* W_hh   = (const float*)d_in[14];  // (192,64) -> used transposed
    const float* b_ih   = (const float*)d_in[15];
    const float* b_hh   = (const float*)d_in[16];
    const float* Wout   = (const float*)d_in[17];  // (64,64)
    const float* bout   = (const float*)d_in[18];

    const int N = in_sizes[0] / 128;
    const size_t sN = (size_t)N;

    // workspace layout (floats), peak = N*452 floats (~90 MB), with reuse
    float* ws    = (float*)d_ws;
    float* ws_z  = ws;                    // N*64
    float* ws_Wh = ws_z  + sN * 64;       // N*128
    float* ws_el = ws_Wh + sN * 128;      // N*2
    float* ws_er = ws_el + sN * 2;        // N*2
    float* ws_xg = ws_er + sN * 2;        // N*256  ([x | gated])
    float* ws_meso = ws_z;                // N*64   (z dead after edge kernel)
    float* ws_gi   = ws_z + sN * 64;      // N*192  (Wh/el/er dead, xg head dead)
    float* ws_gh   = ws_gi + sN * 192;    // N*192  (xg tail dead)

    float* out_src  = (float*)d_out;      // source_term (N,64)
    float* out_newh = out_src + sN * 64;  // new_hidden  (N,64)

    const dim3 blk(256);
    const unsigned rt = (unsigned)((N + 63) / 64);

    // z = x @ Wm + bm                              (K=128, M=64)
    gemm_bias_kernel<<<dim3(rt, 1), blk, 0, stream>>>(x, Wm, bm, ws_z, N, 128, 64, 0);
    // Wh = x @ Wfc                                 (K=128, M=128)
    gemm_bias_kernel<<<dim3(rt, 2), blk, 0, stream>>>(x, Wfc, nullptr, ws_Wh, N, 128, 128, 0);
    // el / er einsums
    el_er_kernel<<<dim3((N + 255) / 256), blk, 0, stream>>>(ws_Wh, attn_l, attn_r, ws_el, ws_er, N);
    // per-node attention + aggregation + gate -> [x | gated]
    edge_agg_kernel<<<dim3((N + 7) / 8), blk, 0, stream>>>(x, ws_z, ws_Wh, ws_el, ws_er,
                                                           src, Wg, bg, ws_xg, N);
    // meso = [x | gated] @ Wmerge + bmerge         (K=256, M=64)
    gemm_bias_kernel<<<dim3(rt, 1), blk, 0, stream>>>(ws_xg, Wmerge, bmerge, ws_meso, N, 256, 64, 0);
    // gi = meso @ W_ih^T + b_ih                    (K=64, M=192, transB)
    gemm_bias_kernel<<<dim3(rt, 3), blk, 0, stream>>>(ws_meso, W_ih, b_ih, ws_gi, N, 64, 192, 1);
    // gh = h @ W_hh^T + b_hh                       (K=64, M=192, transB)
    gemm_bias_kernel<<<dim3(rt, 3), blk, 0, stream>>>(h0, W_hh, b_hh, ws_gh, N, 64, 192, 1);
    // GRU elementwise -> new hidden (second output)
    gru_kernel<<<dim3((unsigned)((sN * 64 + 255) / 256)), blk, 0, stream>>>(ws_gi, ws_gh, h0, out_newh, N);
    // source_term = newh @ Wout + bout             (K=64, M=64)
    gemm_bias_kernel<<<dim3(rt, 1), blk, 0, stream>>>(out_newh, Wout, bout, out_src, N, 64, 64, 0);
}
